// FullAttention_16011638079505
// MI455X (gfx1250) — matrix operand order repaired and checked by reading, not compiled, not run on hardware
//
#include <hip/hip_runtime.h>
#include <hip/hip_bf16.h>

typedef __attribute__((ext_vector_type(16))) _Float16 v16h;
typedef __attribute__((ext_vector_type(8)))  _Float16 v8h;
typedef __attribute__((ext_vector_type(8)))  float    v8f;

#define LOG2E 1.4426950408889634f
#define LNEG  (-23.025850929940457f)   /* -log(1e10), matches reference */

static __device__ __forceinline__ v8f wmma16(v16h a, v16h b, v8f c) {
  // D = A(16x32 f16) * B(32x16 f16) + C(16x16 f32)
  return __builtin_amdgcn_wmma_f32_16x16x32_f16(false, a, false, b, (short)0, c, false, false);
}

// ---------- f32 -> f16 frag builders (used for Q, and by the fallback kernel) ----------
// A-operand layout (16x32, 16-bit): lanes 0-15 hold K {0..7,16..23}, lanes 16-31 hold K {8..15,24..31}
static __device__ __forceinline__ v16h load_a_frag(const float* __restrict__ row, int ebase, int hi) {
  const float4* p0 = (const float4*)(row + ebase + hi * 8);
  const float4* p1 = (const float4*)(row + ebase + 16 + hi * 8);
  float4 a = p0[0], b = p0[1], c = p1[0], d = p1[1];
  v16h r;
  r[0]=(_Float16)a.x;  r[1]=(_Float16)a.y;  r[2]=(_Float16)a.z;  r[3]=(_Float16)a.w;
  r[4]=(_Float16)b.x;  r[5]=(_Float16)b.y;  r[6]=(_Float16)b.z;  r[7]=(_Float16)b.w;
  r[8]=(_Float16)c.x;  r[9]=(_Float16)c.y;  r[10]=(_Float16)c.z; r[11]=(_Float16)c.w;
  r[12]=(_Float16)d.x; r[13]=(_Float16)d.y; r[14]=(_Float16)d.z; r[15]=(_Float16)d.w;
  return r;
}
// B-frag in contiguous k order (32x16, 16-bit): lanes 0-15 hold K 0..15 sequential, lanes 16-31 hold K 16..31
static __device__ __forceinline__ v16h load_b_frag(const float* __restrict__ row, int ebase, int hi) {
  const float4* p = (const float4*)(row + ebase + hi * 16);
  float4 a = p[0], b = p[1], c = p[2], d = p[3];
  v16h r;
  r[0]=(_Float16)a.x;  r[1]=(_Float16)a.y;  r[2]=(_Float16)a.z;  r[3]=(_Float16)a.w;
  r[4]=(_Float16)b.x;  r[5]=(_Float16)b.y;  r[6]=(_Float16)b.z;  r[7]=(_Float16)b.w;
  r[8]=(_Float16)c.x;  r[9]=(_Float16)c.y;  r[10]=(_Float16)c.z; r[11]=(_Float16)c.w;
  r[12]=(_Float16)d.x; r[13]=(_Float16)d.y; r[14]=(_Float16)d.z; r[15]=(_Float16)d.w;
  return r;
}
// A-frag straight from preconverted f16 rows: two aligned 16B loads, zero cvt
static __device__ __forceinline__ v16h load_a_h(const _Float16* __restrict__ row, int ebase, int hi) {
  const v8h* p0 = (const v8h*)(row + ebase + hi * 8);
  const v8h* p1 = (const v8h*)(row + ebase + 16 + hi * 8);
  v8h lo = p0[0], hh = p1[0];
  return __builtin_shufflevector(lo, hh, 0,1,2,3,4,5,6,7,8,9,10,11,12,13,14,15);
}

// ================= pre-pass 1: K f32[b][s][h][e] -> f16 Kh[b][h][s][e] =================
__global__ __launch_bounds__(256) void cvt_k_kernel(const float* __restrict__ K,
                                                    _Float16* __restrict__ Kh) {
  // 524288 threads, 8 elements each; rid enumerates (b,h,s)
  const int t   = blockIdx.x * 256 + threadIdx.x;
  const int rid = t >> 3;
  const int e   = (t & 7) * 8;
  const int b = rid >> 14, h = (rid >> 11) & 7, s = rid & 2047;
  const float4* src = (const float4*)(K + (((size_t)b * 2048 + s) * 8 + h) * 64 + e);
  float4 a = src[0], c = src[1];
  v8h o;
  o[0]=(_Float16)a.x; o[1]=(_Float16)a.y; o[2]=(_Float16)a.z; o[3]=(_Float16)a.w;
  o[4]=(_Float16)c.x; o[5]=(_Float16)c.y; o[6]=(_Float16)c.z; o[7]=(_Float16)c.w;
  *(v8h*)(Kh + (size_t)rid * 64 + e) = o;
}

// ============ pre-pass 2: V f32[b][s][h][d] -> f16 transposed VT[b][h][d][s] ============
__global__ __launch_bounds__(256) void cvt_vt_kernel(const float* __restrict__ V,
                                                     _Float16* __restrict__ VT) {
  // 524288 threads; t = (((b*8+h)*64 + d)*256 + sc), thread handles 8 consecutive s
  const int t  = blockIdx.x * 256 + threadIdx.x;
  const int sc = t & 255;
  const int d  = (t >> 8) & 63;
  const int bh = t >> 14;             // b*8 + h
  const int b  = bh >> 3;
  const int h  = bh & 7;
  const int s0 = sc * 8;
  v8h o;
#pragma unroll
  for (int i = 0; i < 8; ++i) {
    const int s = s0 + i;
    o[i] = (_Float16)V[(((size_t)b * 2048 + s) * 8 + h) * 64 + d];
  }
  *(v8h*)(VT + ((size_t)bh * 64 + d) * 2048 + s0) = o;
}

// ======================= main flash-attention kernel (f16 operands) =====================
// B=4, L=S=2048, H=8, E=D=64. Block = 128 threads (4 wave32), 64 queries/block. No LDS.
__global__ __launch_bounds__(128) void fa_fwd_f16_kernel(
    const float* __restrict__ Q, const _Float16* __restrict__ Kh,
    const _Float16* __restrict__ VT, const int* __restrict__ MSK,
    float* __restrict__ O) {
  constexpr int L = 2048, S = 2048, H = 8, E = 64, D = 64;
  const int bid  = blockIdx.x;
  const int qb   = (bid & 31) * 64;
  const int h    = (bid >> 5) & 7;
  const int b    = bid >> 8;
  const int lane = threadIdx.x & 31;
  const int wave = threadIdx.x >> 5;
  const int r    = lane & 15;
  const int hi   = lane >> 4;
  const int qg   = qb + wave * 16 + r;   // lane = query in score tiles

  const float* qrow = Q + (((size_t)b * L + qg) * H + h) * E;
  const v16h qf0 = load_a_frag(qrow, 0, hi);
  const v16h qf1 = load_a_frag(qrow, 32, hi);

  const int* mrow = MSK + ((size_t)b * L + qg) * S;
  const _Float16* kpane  = Kh + (size_t)(b * H + h) * S * E;
  const _Float16* vtpane = VT + (size_t)(b * H + h) * D * S;

  v8f acc0 = {}, acc1 = {}, acc2 = {}, acc3 = {};
  float mrun = -__builtin_inff();
  float lrun = 0.0f;
  const float sl = 0.125f * LOG2E;   // scale * log2(e)

  for (int kb = 0; kb < S; kb += 32) {
    // ---- scores^T: A = K rows (f16, direct), B = Q ----
    const _Float16* kr0 = kpane + (size_t)(kb + r) * E;
    const _Float16* kr1 = kr0 + 16 * E;
    v16h k00 = load_a_h(kr0, 0, hi), k01 = load_a_h(kr0, 32, hi);
    v16h k10 = load_a_h(kr1, 0, hi), k11 = load_a_h(kr1, 32, hi);
    v8f z = {};
    v8f c0 = wmma16(k00, qf0, wmma16(k01, qf1, z));   // keys kb+0..15
    v8f c1 = wmma16(k10, qf0, wmma16(k11, qf1, z));   // keys kb+16..31

    // prefetch next chunk (global_prefetch_b8)
    __builtin_prefetch(kr0 + 32 * E, 0, 1);
    __builtin_prefetch(vtpane + (size_t)(lane * 2) * S + kb + 32, 0, 1);

    // ---- mask + scale: tile0 keys kb+8*hi+j, tile1 +16 ----
    const int4* mp0 = (const int4*)(mrow + kb + 8 * hi);
    const int4* mp1 = (const int4*)(mrow + kb + 16 + 8 * hi);
    int4 ma = mp0[0], mb = mp0[1], mc = mp1[0], md = mp1[1];
    float t[16];
    t[0]  = (ma.x ? c0[0] : LNEG) * sl;  t[1]  = (ma.y ? c0[1] : LNEG) * sl;
    t[2]  = (ma.z ? c0[2] : LNEG) * sl;  t[3]  = (ma.w ? c0[3] : LNEG) * sl;
    t[4]  = (mb.x ? c0[4] : LNEG) * sl;  t[5]  = (mb.y ? c0[5] : LNEG) * sl;
    t[6]  = (mb.z ? c0[6] : LNEG) * sl;  t[7]  = (mb.w ? c0[7] : LNEG) * sl;
    t[8]  = (mc.x ? c1[0] : LNEG) * sl;  t[9]  = (mc.y ? c1[1] : LNEG) * sl;
    t[10] = (mc.z ? c1[2] : LNEG) * sl;  t[11] = (mc.w ? c1[3] : LNEG) * sl;
    t[12] = (md.x ? c1[4] : LNEG) * sl;  t[13] = (md.y ? c1[5] : LNEG) * sl;
    t[14] = (md.z ? c1[6] : LNEG) * sl;  t[15] = (md.w ? c1[7] : LNEG) * sl;

    // ---- online softmax (per query = lane pair {r, r+16}) ----
    float cm = t[0];
#pragma unroll
    for (int i = 1; i < 16; ++i) cm = fmaxf(cm, t[i]);
    cm = fmaxf(cm, __shfl_xor(cm, 16, 32));
    const float mnew  = fmaxf(mrun, cm);
    const float alpha = __builtin_exp2f(mrun - mnew);
    mrun = mnew;

    v16h pf;                 // P already in the A-operand layout for P*V
    float ps = 0.0f;
#pragma unroll
    for (int i = 0; i < 16; ++i) {
      float p = __builtin_exp2f(t[i] - mnew);
      ps += p;
      pf[i] = (_Float16)p;
    }
    ps += __shfl_xor(ps, 16, 32);
    lrun = lrun * alpha + ps;

    // ---- rescale accumulators: out-acc query = VGPR j + 8*hi ----
    float av[8];
#pragma unroll
    for (int j = 0; j < 8; ++j) av[j] = __shfl(alpha, 8 * hi + j, 32);
#pragma unroll
    for (int j = 0; j < 8; ++j) {
      acc0[j] *= av[j]; acc1[j] *= av[j]; acc2[j] *= av[j]; acc3[j] *= av[j];
    }

    // ---- P(16q x 32k) * V(32k x 64d): B-frags direct from transposed f16 V ----
#pragma unroll
    for (int tile = 0; tile < 4; ++tile) {
      const v8h* vp = (const v8h*)(vtpane + (size_t)(tile * 16 + r) * S + kb + hi * 8);
      v8h lo = vp[0], hh = vp[2];
      v16h vf = __builtin_shufflevector(lo, hh, 0,1,2,3,4,5,6,7,8,9,10,11,12,13,14,15);
      if (tile == 0) acc0 = wmma16(pf, vf, acc0);
      if (tile == 1) acc1 = wmma16(pf, vf, acc1);
      if (tile == 2) acc2 = wmma16(pf, vf, acc2);
      if (tile == 3) acc3 = wmma16(pf, vf, acc3);
    }
  }

  // ---- epilogue: divide by row sum; query = j + 8*hi, d = tile*16 + r ----
  const float linv = 1.0f / lrun;
  float lv[8];
#pragma unroll
  for (int j = 0; j < 8; ++j) lv[j] = __shfl(linv, 8 * hi + j, 32);
#pragma unroll
  for (int j = 0; j < 8; ++j) {
    const int q = qb + wave * 16 + 8 * hi + j;
    float* op = O + (((size_t)b * L + q) * H + h) * D + r;
    op[0]  = acc0[j] * lv[j];
    op[16] = acc1[j] * lv[j];
    op[32] = acc2[j] * lv[j];
    op[48] = acc3[j] * lv[j];
  }
}

// =============== fallback: fully inline version (no workspace needed) ===============
__global__ __launch_bounds__(128) void fa_fwd_f32_kernel(
    const float* __restrict__ Q, const float* __restrict__ K,
    const float* __restrict__ V, const int* __restrict__ MSK,
    float* __restrict__ O) {
  constexpr int L = 2048, S = 2048, H = 8, E = 64, D = 64;
  __shared__ __align__(16) _Float16 vt[2][64][40];
  const int bid  = blockIdx.x;
  const int qb   = (bid & 31) * 64;
  const int h    = (bid >> 5) & 7;
  const int b    = bid >> 8;
  const int tid  = threadIdx.x;
  const int lane = tid & 31, wave = tid >> 5;
  const int r = lane & 15, hi = lane >> 4;
  const int qg = qb + wave * 16 + r;

  const float* qrow = Q + (((size_t)b * L + qg) * H + h) * E;
  const v16h qf0 = load_a_frag(qrow, 0, hi);
  const v16h qf1 = load_a_frag(qrow, 32, hi);
  const int* mrow = MSK + ((size_t)b * L + qg) * S;

  v8f acc0 = {}, acc1 = {}, acc2 = {}, acc3 = {};
  float mrun = -__builtin_inff(), lrun = 0.0f;
  const float sl = 0.125f * LOG2E;
  const int vk = tid & 31, vd = (tid >> 5) * 16;

  for (int kb = 0; kb < S; kb += 32) {
    const int buf = (kb >> 5) & 1;
    {
      const float* vrow = V + (((size_t)b * S + kb + vk) * H + h) * D + vd;
      const float4* vp = (const float4*)vrow;
      float4 f0 = vp[0], f1 = vp[1], f2 = vp[2], f3 = vp[3];
      float fv[16] = { f0.x,f0.y,f0.z,f0.w, f1.x,f1.y,f1.z,f1.w,
                       f2.x,f2.y,f2.z,f2.w, f3.x,f3.y,f3.z,f3.w };
#pragma unroll
      for (int j = 0; j < 16; ++j) vt[buf][vd + j][vk] = (_Float16)fv[j];
    }
    __syncthreads();

    const float* krow0 = K + (((size_t)b * S + kb + r) * H + h) * E;
    const float* krow1 = krow0 + (size_t)16 * H * E;
    v16h k00 = load_a_frag(krow0, 0, hi), k01 = load_a_frag(krow0, 32, hi);
    v16h k10 = load_a_frag(krow1, 0, hi), k11 = load_a_frag(krow1, 32, hi);
    v8f z = {};
    v8f c0 = wmma16(k00, qf0, wmma16(k01, qf1, z));
    v8f c1 = wmma16(k10, qf0, wmma16(k11, qf1, z));

    const int4* mp0 = (const int4*)(mrow + kb + 8 * hi);
    const int4* mp1 = (const int4*)(mrow + kb + 16 + 8 * hi);
    int4 ma = mp0[0], mb = mp0[1], mc = mp1[0], md = mp1[1];
    float t[16];
    t[0]  = (ma.x ? c0[0] : LNEG) * sl;  t[1]  = (ma.y ? c0[1] : LNEG) * sl;
    t[2]  = (ma.z ? c0[2] : LNEG) * sl;  t[3]  = (ma.w ? c0[3] : LNEG) * sl;
    t[4]  = (mb.x ? c0[4] : LNEG) * sl;  t[5]  = (mb.y ? c0[5] : LNEG) * sl;
    t[6]  = (mb.z ? c0[6] : LNEG) * sl;  t[7]  = (mb.w ? c0[7] : LNEG) * sl;
    t[8]  = (mc.x ? c1[0] : LNEG) * sl;  t[9]  = (mc.y ? c1[1] : LNEG) * sl;
    t[10] = (mc.z ? c1[2] : LNEG) * sl;  t[11] = (mc.w ? c1[3] : LNEG) * sl;
    t[12] = (md.x ? c1[4] : LNEG) * sl;  t[13] = (md.y ? c1[5] : LNEG) * sl;
    t[14] = (md.z ? c1[6] : LNEG) * sl;  t[15] = (md.w ? c1[7] : LNEG) * sl;

    float cm = t[0];
#pragma unroll
    for (int i = 1; i < 16; ++i) cm = fmaxf(cm, t[i]);
    cm = fmaxf(cm, __shfl_xor(cm, 16, 32));
    const float mnew  = fmaxf(mrun, cm);
    const float alpha = __builtin_exp2f(mrun - mnew);
    mrun = mnew;

    v16h pf; float ps = 0.0f;
#pragma unroll
    for (int i = 0; i < 16; ++i) {
      float p = __builtin_exp2f(t[i] - mnew);
      ps += p; pf[i] = (_Float16)p;
    }
    ps += __shfl_xor(ps, 16, 32);
    lrun = lrun * alpha + ps;

    float av[8];
#pragma unroll
    for (int j = 0; j < 8; ++j) av[j] = __shfl(alpha, 8 * hi + j, 32);
#pragma unroll
    for (int j = 0; j < 8; ++j) {
      acc0[j] *= av[j]; acc1[j] *= av[j]; acc2[j] *= av[j]; acc3[j] *= av[j];
    }
#pragma unroll
    for (int tile = 0; tile < 4; ++tile) {
      const v8h* lp = (const v8h*)&vt[buf][tile * 16 + r][hi * 8];
      v8h lo = lp[0], hh = lp[2];
      v16h vf = __builtin_shufflevector(lo, hh, 0,1,2,3,4,5,6,7,8,9,10,11,12,13,14,15);
      if (tile == 0) acc0 = wmma16(pf, vf, acc0);
      if (tile == 1) acc1 = wmma16(pf, vf, acc1);
      if (tile == 2) acc2 = wmma16(pf, vf, acc2);
      if (tile == 3) acc3 = wmma16(pf, vf, acc3);
    }
  }

  const float linv = 1.0f / lrun;
  float lv[8];
#pragma unroll
  for (int j = 0; j < 8; ++j) lv[j] = __shfl(linv, 8 * hi + j, 32);
#pragma unroll
  for (int j = 0; j < 8; ++j) {
    const int q = qb + wave * 16 + 8 * hi + j;
    float* op = O + (((size_t)b * L + q) * H + h) * D + r;
    op[0]  = acc0[j] * lv[j];
    op[16] = acc1[j] * lv[j];
    op[32] = acc2[j] * lv[j];
    op[48] = acc3[j] * lv[j];
  }
}

extern "C" void kernel_launch(void* const* d_in, const int* in_sizes, int n_in,
                              void* d_out, int out_size, void* d_ws, size_t ws_size,
                              hipStream_t stream) {
  (void)in_sizes; (void)n_in; (void)out_size;
  const float* Q = (const float*)d_in[0];
  const float* K = (const float*)d_in[1];
  const float* V = (const float*)d_in[2];
  const int*   M = (const int*)d_in[3];
  float*       O = (float*)d_out;

  const size_t needK = (size_t)4 * 8 * 2048 * 64 * sizeof(_Float16);  // 8.39 MB
  const size_t needV = needK;
  if (d_ws != nullptr && ws_size >= needK + needV) {
    _Float16* Kh = (_Float16*)d_ws;
    _Float16* VT = (_Float16*)((char*)d_ws + needK);
    cvt_k_kernel <<<dim3(2048), dim3(256), 0, stream>>>(K, Kh);
    cvt_vt_kernel<<<dim3(2048), dim3(256), 0, stream>>>(V, VT);
    fa_fwd_f16_kernel<<<dim3(1024), dim3(128), 0, stream>>>(Q, Kh, VT, M, O);
  } else {
    fa_fwd_f32_kernel<<<dim3(1024), dim3(128), 0, stream>>>(Q, K, V, M, O);
  }
}
